// EARLIEST_53317724012884
// MI455X (gfx1250) — matrix operand order repaired and checked by reading, not compiled, not run on hardware
//
#include <hip/hip_runtime.h>
#include <hip/hip_bf16.h>

// ---------------- problem constants ----------------
#define B_    128
#define T_    4096
#define V_    128
#define H_    256
#define C_    10
#define G4H   1024
#define EPS_  0.1f

// ---------------- partitioning --------------------
#define NSLICE 8                 // N-slice workgroups per batch group
#define MG     16                // batch rows per group (one 16-row WMMA tile)
#define NG     (B_ / MG)         // 8 batch groups
#define KTOT   (V_ + H_)         // 384
#define KT     (KTOT / 32)       // 12 k-tiles of 32
#define NLOC   128               // local gate columns per WG (4 gates x 32 h cols)
#define HS     32                // h columns owned by one WG slice

typedef __bf16 v16bf __attribute__((ext_vector_type(16)));
typedef __bf16 v8bf  __attribute__((ext_vector_type(8)));
typedef __bf16 v4bf  __attribute__((ext_vector_type(4)));
typedef __bf16 v2bf  __attribute__((ext_vector_type(2)));
typedef float  v8f   __attribute__((ext_vector_type(8)));
typedef float  v4f   __attribute__((ext_vector_type(4)));

__device__ __forceinline__ float sigm(float x) { return 1.0f / (1.0f + __expf(-x)); }
__device__ __forceinline__ float ftanh(float x) {
  const float e = __expf(2.0f * x);
  return (e - 1.0f) / (e + 1.0f);
}

// ---------------------------------------------------------------------------
// workspace init: zero h double-buffer + per-group barrier counters
// ---------------------------------------------------------------------------
__global__ void init_ws(unsigned* __restrict__ bar, unsigned short* __restrict__ hbuf) {
  const int i = blockIdx.x * blockDim.x + threadIdx.x;
  if (i < NG) bar[i] = 0u;
  const int tot = 2 * NG * MG * H_;
  for (int j = i; j < tot; j += gridDim.x * blockDim.x) hbuf[j] = 0;
}

// logits = softmax(h @ Wo + bo), into lbuf (normalized). Uniform-branch safe.
__device__ __forceinline__ void logits_softmax(const __bf16* hb, const float* lWo,
                                               const float* lbo, float* lbuf, int tid) {
  for (int i = tid; i < MG * C_; i += 256) {
    const int m = i / C_, cl = i - m * C_;
    float acc = lbo[cl];
#pragma unroll 8
    for (int j = 0; j < H_; ++j) acc += (float)hb[m * H_ + j] * lWo[j * C_ + cl];
    lbuf[i] = acc;
  }
  __syncthreads();
  if (tid < MG) {
    const int m = tid;
    float mx = lbuf[m * C_];
#pragma unroll
    for (int c = 1; c < C_; ++c) mx = fmaxf(mx, lbuf[m * C_ + c]);
    float e[C_], sum = 0.0f;
#pragma unroll
    for (int c = 0; c < C_; ++c) { e[c] = __expf(lbuf[m * C_ + c] - mx); sum += e[c]; }
    const float inv = 1.0f / sum;
#pragma unroll
    for (int c = 0; c < C_; ++c) lbuf[m * C_ + c] = e[c] * inv;
  }
  __syncthreads();
}

// ---------------------------------------------------------------------------
// persistent recurrent kernel: 64 WGs (8 batch groups x 8 N slices), wave32 WMMA
// ---------------------------------------------------------------------------
__global__ __launch_bounds__(256, 1) void earliest_rnn(
    const float* __restrict__ X, const float* __restrict__ U,
    const float* __restrict__ Wk, const float* __restrict__ Wr,
    const float* __restrict__ bl, const float* __restrict__ Wo,
    const float* __restrict__ bo, const float* __restrict__ Wc,
    const float* __restrict__ bc, float* __restrict__ out,
    __bf16* __restrict__ hbuf, unsigned* __restrict__ bar) {
  __shared__ __attribute__((aligned(16))) __bf16 Abuf[MG * KTOT];  // [x_t | h] row-major
  __shared__ float zbuf[MG * NLOC];   // per-slice gate pre-activations
  __shared__ float cbuf[MG * HS];     // persistent cell state slice
  __shared__ float preds[MG * C_];    // halting predictions (slice 0)
  __shared__ int   halted[MG];
  __shared__ float lbuf[MG * C_];
  __shared__ float lWo[H_ * C_];
  __shared__ float lWc[H_ + 1];
  __shared__ float lbo[C_];

  const int tid   = threadIdx.x;
  const int lane  = tid & 31;
  const int wav   = tid >> 5;              // 0..7
  const int s     = blockIdx.x % NSLICE;   // N slice
  const int g     = blockIdx.x / NSLICE;   // batch group
  const int khalf = lane >> 4;             // 0/1: K-half per ISA frag layout

  // gate-column mapping: local col -> global gate col (i|f|g|o interleaved by slice)
  const int nloc   = wav * 16 + (lane & 15);
  const int gate   = nloc >> 5;
  const int within = nloc & 31;
  const int nglob  = gate * H_ + s * HS + within;

  // ---- preload B fragments (weights) into VGPRs, bf16, ISA B-layout ----
  v16bf bfrag[KT];
#pragma unroll
  for (int kt = 0; kt < KT; ++kt) {
#pragma unroll
    for (int j = 0; j < 16; ++j) {
      const int k = kt * 32 + ((j < 8) ? (khalf * 8 + j) : (16 + khalf * 8 + (j - 8)));
      const float w = (k < V_) ? Wk[k * G4H + nglob] : Wr[(k - V_) * G4H + nglob];
      bfrag[kt][j] = (__bf16)w;
    }
  }
  const float bv = bl[nglob];

  for (int i = tid; i < MG * HS; i += 256) cbuf[i] = 0.0f;
  if (s == 0) {
    for (int i = tid; i < MG * C_; i += 256) preds[i] = 0.0f;
    if (tid < MG) halted[tid] = 0;
    for (int i = tid; i < H_ * C_; i += 256) lWo[i] = Wo[i];
    if (tid < H_ + 1) lWc[tid] = Wc[tid];
    if (tid < C_) lbo[tid] = bo[tid];
  }
  __syncthreads();

  const float bc0 = bc[0];

  for (int t = 0; t < T_; ++t) {
    const __bf16* hin  = hbuf + ((size_t)((t)     & 1) * NG + g) * (MG * H_);
    __bf16*       hout = hbuf + ((size_t)((t + 1) & 1) * NG + g) * (MG * H_);

    // ---- assemble A = [x_t | h] in LDS (row-major, bf16), vectorized ----
#pragma unroll
    for (int it = 0; it < (MG * V_) / (256 * 4); ++it) {
      const int i = (it * 256 + tid) * 4;
      const int m = i >> 7, v = i & (V_ - 1);
      const v4f xv = *(const v4f*)&X[((size_t)(g * MG + m)) * T_ * V_ + (size_t)t * V_ + v];
      v4bf xb;
#pragma unroll
      for (int q = 0; q < 4; ++q) xb[q] = (__bf16)xv[q];
      *(v4bf*)&Abuf[m * KTOT + v] = xb;
    }
#pragma unroll
    for (int it = 0; it < (MG * H_) / (256 * 4); ++it) {
      const int i = (it * 256 + tid) * 4;
      const int m = i >> 8, j = i & (H_ - 1);
      *(v4bf*)&Abuf[m * KTOT + V_ + j] = *(const v4bf*)&hin[m * H_ + j];
    }
    if (t + 1 < T_ && tid < MG)  // global_prefetch_b8 for next x slab
      __builtin_prefetch(&X[((size_t)(g * MG + tid)) * T_ * V_ + (size_t)(t + 1) * V_], 0, 1);

    // ---- halting/logits bookkeeping for step t-1 (needs full h) ----
    if (s == 0 && t > 0) {
      logits_softmax(hin, lWo, lbo, lbuf, tid);
      if (tid < MG) {
        const int m = tid;
        float acc = bc0 + (float)(t - 1) * lWc[H_];
#pragma unroll 8
        for (int j = 0; j < H_; ++j) acc += (float)hin[m * H_ + j] * lWc[j];
        const float probs = (1.0f - EPS_) * sigm(acc) + EPS_ * 0.05f;
        const float uu = U[(size_t)(g * MG + m) * T_ + (t - 1)];
        if (!halted[m] && uu < probs) {
          halted[m] = 1;
#pragma unroll
          for (int c = 0; c < C_; ++c) preds[m * C_ + c] = lbuf[m * C_ + c];
        }
      }
    }
    __syncthreads();

    // ---- pipelined z-tile GEMM: ds_read_b128 stream ahead of WMMA chain ----
    const int m = lane & 15;
    v16bf afrag[KT];
#pragma unroll
    for (int kt = 0; kt < KT; ++kt) {
      const v8bf lo = *(const v8bf*)&Abuf[m * KTOT + kt * 32 + khalf * 8];
      const v8bf hi = *(const v8bf*)&Abuf[m * KTOT + kt * 32 + 16 + khalf * 8];
      afrag[kt] = __builtin_shufflevector(lo, hi, 0, 1, 2, 3, 4, 5, 6, 7,
                                          8, 9, 10, 11, 12, 13, 14, 15);
    }
    v8f acc = {0.f, 0.f, 0.f, 0.f, 0.f, 0.f, 0.f, 0.f};
#pragma unroll
    for (int kt = 0; kt < KT; ++kt)
      acc = __builtin_amdgcn_wmma_f32_16x16x32_bf16(false, afrag[kt], false, bfrag[kt],
                                                    (short)0, acc, false, false);

#if __has_builtin(__builtin_amdgcn_sched_group_barrier)
    // Pipeline shape: 4 DS-reads prologue (2 tiles ahead), then 10x[1 WMMA, 2 DS-reads],
    // then the last 2 WMMAs. Keeps <=3 A fragments live; waits become partial dscnt.
    __builtin_amdgcn_sched_group_barrier(0x100, 4, 0);   // 4 DS reads
#pragma unroll
    for (int i = 0; i < 10; ++i) {
      __builtin_amdgcn_sched_group_barrier(0x008, 1, 0); // 1 WMMA
      __builtin_amdgcn_sched_group_barrier(0x100, 2, 0); // 2 DS reads
    }
    __builtin_amdgcn_sched_group_barrier(0x008, 2, 0);   // final 2 WMMAs
#endif

    const int mbase = khalf * 8;  // lanes 16-31 hold rows M=8..15
#pragma unroll
    for (int r = 0; r < 8; ++r) zbuf[(mbase + r) * NLOC + nloc] = acc[r] + bv;
    __syncthreads();

    // ---- gate combine + state update (512 elems, 2 adjacent cols per thread) ----
    {
      const int idx = tid * 2;
      const int mm = idx >> 5, j = idx & 31;  // j even -> j, j+1 in same row
      float hv[2];
#pragma unroll
      for (int e = 0; e < 2; ++e) {
        const float iv = zbuf[mm * NLOC + j + e];
        const float fv = zbuf[mm * NLOC + 32 + j + e];
        const float gv = zbuf[mm * NLOC + 64 + j + e];
        const float ov = zbuf[mm * NLOC + 96 + j + e];
        const float c  = sigm(fv) * cbuf[mm * HS + j + e] + sigm(iv) * ftanh(gv);
        cbuf[mm * HS + j + e] = c;
        hv[e] = sigm(ov) * ftanh(c);
      }
      v2bf hp;
      hp[0] = (__bf16)hv[0];
      hp[1] = (__bf16)hv[1];
      *(v2bf*)&hout[mm * H_ + s * HS + j] = hp;  // one 4-byte global store
    }
    __threadfence();
    __syncthreads();

    // ---- per-group barrier (monotonic counter, deterministic) ----
    if (tid == 0) {
      __hip_atomic_fetch_add(&bar[g], 1u, __ATOMIC_RELEASE, __HIP_MEMORY_SCOPE_AGENT);
      const unsigned tgt = (unsigned)(NSLICE * (t + 1));
      while (__hip_atomic_load(&bar[g], __ATOMIC_ACQUIRE, __HIP_MEMORY_SCOPE_AGENT) < tgt)
        __builtin_amdgcn_s_sleep(1);
    }
    __syncthreads();
  }

  // ---- final: logits at T-1, output = halted ? preds : logits_last ----
  if (s == 0) {
    const __bf16* hfin = hbuf + ((size_t)(T_ & 1) * NG + g) * (MG * H_);
    logits_softmax(hfin, lWo, lbo, lbuf, tid);
    for (int i = tid; i < MG * C_; i += 256) {
      const int m2 = i / C_;
      out[(size_t)(g * MG + m2) * C_ + (i - m2 * C_)] = halted[m2] ? preds[i] : lbuf[i];
    }
  }
}

// ---------------------------------------------------------------------------
extern "C" void kernel_launch(void* const* d_in, const int* in_sizes, int n_in,
                              void* d_out, int out_size, void* d_ws, size_t ws_size,
                              hipStream_t stream) {
  const float* X  = (const float*)d_in[0];
  const float* U  = (const float*)d_in[1];
  const float* Wk = (const float*)d_in[2];
  const float* Wr = (const float*)d_in[3];
  const float* bl = (const float*)d_in[4];
  const float* Wo = (const float*)d_in[5];
  const float* bo = (const float*)d_in[6];
  const float* Wc = (const float*)d_in[7];
  const float* bc = (const float*)d_in[8];
  // d_in[9] (Wb), d_in[10] (bb): baseline head, does not affect returned logits
  float* out = (float*)d_out;

  __bf16*   hbuf = (__bf16*)d_ws;                                  // 2*NG*MG*H_ bf16
  unsigned* bar  = (unsigned*)((char*)d_ws + 2 * NG * MG * H_ * sizeof(__bf16));

  init_ws<<<8, 256, 0, stream>>>(bar, (unsigned short*)d_ws);
  earliest_rnn<<<NG * NSLICE, 256, 0, stream>>>(X, U, Wk, Wr, bl, Wo, bo, Wc, bc,
                                                out, hbuf, bar);
}
